// ArthTextToDenseBlock_2319282340019
// MI455X (gfx1250) — matrix operand order repaired and checked by reading, not compiled, not run on hardware
//
#include <hip/hip_runtime.h>

#define NDIM 512
#define NB   1024
#define NS   512

typedef __bf16 bf16;
typedef __attribute__((ext_vector_type(16))) __bf16 v16bf;
typedef __attribute__((ext_vector_type(8)))  __bf16 v8bf;
typedef __attribute__((ext_vector_type(8)))  float  v8f;

// ---------- helpers ----------
__device__ __forceinline__ bf16 f2bf(float f) {
  union { float f; unsigned u; } a; a.f = f;
  unsigned u = a.u + 0x7FFFu + ((a.u >> 16) & 1u);
  unsigned short h = (unsigned short)(u >> 16);
  return __builtin_bit_cast(bf16, h);
}

// A-fragment (16x32 bf16) per ISA layout:
// lanes 0-15: row=lane, elems = K{0..7} then K{16..23}
// lanes 16-31: row=lane-16, elems = K{8..15} then K{24..31}
__device__ __forceinline__ v16bf lds_a_frag(const bf16* base, int ldk,
                                            int laneM, int hi, int k0) {
  const bf16* row = base + laneM * ldk + k0;
  v8bf lo = *(const v8bf*)(row + hi * 8);
  v8bf hv = *(const v8bf*)(row + 16 + hi * 8);
  return __builtin_shufflevector(lo, hv, 0,1,2,3,4,5,6,7,8,9,10,11,12,13,14,15);
}

#define WMMA_BF16(A, Bv, C) \
  __builtin_amdgcn_wmma_f32_16x16x32_bf16(false, (A), false, (Bv), (short)0, (C), false, false)

// One FF layer over a 64-row tile: out = leaky_relu(in @ Wpacked + bias).
// 16 waves split N; each B-fragment is reused for 4 M-groups (4 accumulators).
// Optional second output with bias+csum (denseop fds=1 variant).
__device__ __forceinline__ void layer_ff64(const bf16* in_lds, int K,
                                           const bf16* wpk, const float* bias,
                                           bf16* out_lds, int N,
                                           const float* csum, bf16* out2, int tid) {
  const int wave = tid >> 5, lane = tid & 31;
  const int laneM = lane & 15, hi = lane >> 4;
  const int nk = K >> 5, ntiles = N >> 4;
  const v8f z = (v8f){0.f,0.f,0.f,0.f,0.f,0.f,0.f,0.f};
  for (int nt = wave; nt < ntiles; nt += 16) {
    v8f acc0 = z, acc1 = z, acc2 = z, acc3 = z;
    const v16bf* bp = reinterpret_cast<const v16bf*>(wpk + (size_t)nt * nk * 512) + lane;
    for (int kt = 0; kt < nk; ++kt) {
      v16bf b = *bp; bp += 32;
      v16bf a0 = lds_a_frag(in_lds,          K, laneM, hi, kt * 32);
      v16bf a1 = lds_a_frag(in_lds + 16 * K, K, laneM, hi, kt * 32);
      v16bf a2 = lds_a_frag(in_lds + 32 * K, K, laneM, hi, kt * 32);
      v16bf a3 = lds_a_frag(in_lds + 48 * K, K, laneM, hi, kt * 32);
      acc0 = WMMA_BF16(a0, b, acc0);
      acc1 = WMMA_BF16(a1, b, acc1);
      acc2 = WMMA_BF16(a2, b, acc2);
      acc3 = WMMA_BF16(a3, b, acc3);
    }
    const int n = nt * 16 + laneM;
    const float bv = bias[n];
#pragma unroll
    for (int r = 0; r < 8; ++r) {
      float v0 = acc0[r] + bv; v0 = v0 > 0.f ? v0 : 0.01f * v0;
      float v1 = acc1[r] + bv; v1 = v1 > 0.f ? v1 : 0.01f * v1;
      float v2 = acc2[r] + bv; v2 = v2 > 0.f ? v2 : 0.01f * v2;
      float v3 = acc3[r] + bv; v3 = v3 > 0.f ? v3 : 0.01f * v3;
      const int rr = r + hi * 8;
      out_lds[(rr)      * N + n] = f2bf(v0);
      out_lds[(rr + 16) * N + n] = f2bf(v1);
      out_lds[(rr + 32) * N + n] = f2bf(v2);
      out_lds[(rr + 48) * N + n] = f2bf(v3);
    }
    if (out2) {
      const float bv2 = bv + csum[n];
#pragma unroll
      for (int r = 0; r < 8; ++r) {
        float v0 = acc0[r] + bv2; v0 = v0 > 0.f ? v0 : 0.01f * v0;
        float v1 = acc1[r] + bv2; v1 = v1 > 0.f ? v1 : 0.01f * v1;
        float v2 = acc2[r] + bv2; v2 = v2 > 0.f ? v2 : 0.01f * v2;
        float v3 = acc3[r] + bv2; v3 = v3 > 0.f ? v3 : 0.01f * v3;
        const int rr = r + hi * 8;
        out2[(rr)      * N + n] = f2bf(v0);
        out2[(rr + 16) * N + n] = f2bf(v1);
        out2[(rr + 32) * N + n] = f2bf(v2);
        out2[(rr + 48) * N + n] = f2bf(v3);
      }
    }
  }
}

// Output layer: logits (cols padded to 16) -> LOG[64][16] f32. Waves 0-3, one M-group each.
__device__ __forceinline__ void layer_out64(const bf16* in_lds, int K,
                                            const bf16* wpk, const float* bias,
                                            int nout, float* LOGp, int tid) {
  const int wave = tid >> 5;
  if (wave >= 4) return;
  const int lane = tid & 31, laneM = lane & 15, hi = lane >> 4;
  const int nk = K >> 5;
  v8f acc = (v8f){0.f,0.f,0.f,0.f,0.f,0.f,0.f,0.f};
  const v16bf* bp = reinterpret_cast<const v16bf*>(wpk) + lane;
  const bf16* in_m = in_lds + wave * 16 * K;
  for (int kt = 0; kt < nk; ++kt) {
    v16bf a = lds_a_frag(in_m, K, laneM, hi, kt * 32);
    v16bf b = *bp; bp += 32;
    acc = WMMA_BF16(a, b, acc);
  }
  const float bv = (laneM < nout) ? bias[laneM] : 0.f;
#pragma unroll
  for (int r = 0; r < 8; ++r)
    LOGp[(wave * 16 + r + hi * 8) * 16 + laneM] = acc[r] + bv;
}

// ---------- prep kernels ----------
// Pack W (K x Nsrc row-major f32) into bf16 B-fragment tiles (Npad cols).
// Tile (kt,nt) base = (nt*nk + kt)*512; elem = lane*16 + j with
// col = nt*16 + (lane&15), k = kt*32 + (lane>>4)*8 + (j&7) + (j>>3)*16.
__global__ void pack_w(const float* __restrict__ W, bf16* __restrict__ dst,
                       int K, int Npad, int Nsrc) {
  int idx = blockIdx.x * blockDim.x + threadIdx.x;
  int nk = K >> 5;
  int total = (Npad >> 4) * nk * 512;
  if (idx >= total) return;
  int tile = idx >> 9, e = idx & 511;
  int lane = e >> 4, j = e & 15;
  int nt = tile / nk, kt = tile - nt * nk;
  int n = nt * 16 + (lane & 15);
  int k = kt * 32 + (lane >> 4) * 8 + (j & 7) + ((j >> 3) << 4);
  float v = (n < Nsrc) ? W[(size_t)k * Nsrc + n] : 0.f;
  dst[idx] = f2bf(v);
}

// csum[n] = sum_{k=512..1023} W1_denseop[k][n]  (rank-1 fds term)
__global__ void csum_kernel(const float* __restrict__ W1d, float* __restrict__ c) {
  int n = blockIdx.x * blockDim.x + threadIdx.x;
  if (n >= NDIM) return;
  float s = 0.f;
  for (int k = 512; k < 1024; ++k) s += W1d[(size_t)k * NDIM + n];
  c[n] = s;
}

// ---------- gates kernel ----------
struct GParams {
  const float* b1[6];
  const float* b2[6];
  const float* b3[6];
  const float* csum;
  float* gatesf;           // [t*NB+b][8] : op_pred[0..6], digit
  unsigned int* gflags;    // bit0 ignore, bit1 move2, bits2-3 gate0, bits4-5 gate1, bit6 dsg
};

// Dynamic LDS: Xs 64x512 | H1a 64x512 | H1b 64x512 | H2 64x256 | LOG 64x16 f32 | uflags 64
#define SMEM_BYTES (65536 + 65536 + 65536 + 32768 + 4096 + 256)

__global__ __launch_bounds__(512) void gates_kernel(const float* __restrict__ x,
                                                    const bf16* __restrict__ pw,
                                                    GParams P) {
  extern __shared__ __align__(16) char smem[];
  bf16* Xs  = (bf16*)smem;
  bf16* H1a = Xs + 64 * 512;
  bf16* H1b = H1a + 64 * 512;
  bf16* H2  = H1b + 64 * 512;
  float* LOG = (float*)(H2 + 64 * 256);
  unsigned int* uflags = (unsigned int*)(LOG + 64 * 16);

  const int tid = threadIdx.x;
  const long rowbase = (long)blockIdx.x * 64;

  for (int i = tid; i < 64 * 512; i += 512)
    Xs[i] = f2bf(x[rowbase * 512 + i]);
  if (tid < 64) uflags[tid] = 0u;
  __syncthreads();

  const int OUTS[6] = {2, 3, 7, 2, 10, 4};  // valid, move, op, dec, densepred, denseop

  for (int m = 0; m < 6; ++m) {
    const bf16* wm = pw + (size_t)m * 397312;
    const bool dual = (m == 5);
    layer_ff64(Xs, 512, wm, P.b1[m], H1a, 512,
               dual ? P.csum : nullptr, dual ? H1b : nullptr, tid);
    __syncthreads();
    const int nvar = dual ? 2 : 1;
    for (int v = 0; v < nvar; ++v) {
      layer_ff64(v ? H1b : H1a, 512, wm + 262144, P.b2[m], H2, 256, nullptr, nullptr, tid);
      __syncthreads();
      layer_out64(H2, 256, wm + 393216, P.b3[m], OUTS[m], LOG, tid);
      __syncthreads();
      if (tid < 64) {
        const float* row = LOG + tid * 16;
        long g = rowbase + tid;
        int bI = (int)(g >> 9), tI = (int)(g & 511);
        size_t gidx = (size_t)tI * NB + (size_t)bI;
        int nout = OUTS[m];
        int bi = 0; float bv = row[0];
        for (int j = 1; j < nout; ++j) { float vv = row[j]; if (vv > bv) { bv = vv; bi = j; } }
        if (m == 0)      { if (bi == 1) uflags[tid] |= 1u; }
        else if (m == 1) { if (bi == 2) uflags[tid] |= 2u; }
        else if (m == 2) {
          float s = 0.f;
          for (int j = 0; j < 7; ++j) s += row[j] * row[j];
          float inv = 1.f / fmaxf(sqrtf(s), 1e-5f);
          for (int j = 0; j < 7; ++j) P.gatesf[gidx * 8 + j] = row[j] * inv;
        }
        else if (m == 3) { if (bi == 1) uflags[tid] |= 64u; }
        else if (m == 4) { P.gatesf[gidx * 8 + 7] = (float)bi; }
        else             { uflags[tid] |= ((unsigned)bi) << (v ? 4 : 2); }
      }
      __syncthreads();
    }
  }
  if (tid < 64) {
    long g = rowbase + tid;
    int bI = (int)(g >> 9), tI = (int)(g & 511);
    P.gflags[(size_t)tI * NB + bI] = uflags[tid];
  }
}

// ---------- scan kernel: one thread per batch element ----------
__global__ void scan_kernel(const float* __restrict__ gatesf,
                            const unsigned int* __restrict__ gflags,
                            float* __restrict__ out,
                            const int* __restrict__ start_pos_p) {
  int b = blockIdx.x * blockDim.x + threadIdx.x;
  if (b >= NB) return;
  const int t0 = *start_pos_p;
  float* od = out + (size_t)b * 4608;  // trans_dense[512]
  float* ov = od + 512;                // trans_valid[512]
  float* oo = od + 1024;               // trans_op[512][7]
  int p = 0;
  float cur = 0.f, vcur = 0.f, fpm = 1.f, fds = 0.f;
  for (int t = t0; t < NS; ++t) {
    size_t idx = (size_t)t * NB + b;
    unsigned int u = gflags[idx];
    const float* gf = gatesf + idx * 8;
    float ign = (float)(u & 1u);
    float mv2 = (float)((u >> 1) & 1u);
    float g_move = (1.f - ign) * mv2;
    float dsg = (float)((u >> 6) & 1u);
    int gi = (fds > 0.5f) ? (int)((u >> 4) & 3u) : (int)((u >> 2) & 3u);
    float wg = 1.f - ign;
    float iw = wg * (gi == 1 ? 1.f : 0.f);
    float dw = wg * (gi == 2 ? 1.f : 0.f);
    float digit = gf[7];
    if (g_move > 0.5f) {                 // pos_mask shift (before reads, per reference)
      if (p < 512) { od[p] = cur; ov[p] = vcur; }
      ++p; cur = 0.f; vcur = 0.f;
      if (p < 512) {
#pragma unroll
        for (int j = 0; j < 7; ++j) oo[p * 7 + j] = gf[j];  // op_pred written at arrival
      }
    }
    if (p < 512) {
      float nv = iw * (cur * 10.f + digit) + dw * (cur + digit * fpm * 0.1f)
               + (1.f - iw - dw) * cur;
      cur = nv;
      vcur = fmaxf(vcur, iw + dw);
    }
    fpm = dw * (fpm * 0.1f) + g_move + (1.f - dw) * (1.f - g_move) * fpm;
    fds = (1.f - g_move) * fmaxf(fds, dsg);
  }
  if (p < 512) { od[p] = cur; ov[p] = vcur; }
}

// ---------- launcher ----------
extern "C" void kernel_launch(void* const* d_in, const int* in_sizes, int n_in,
                              void* d_out, int out_size, void* d_ws, size_t ws_size,
                              hipStream_t stream) {
  (void)in_sizes; (void)n_in;
  const float* x = (const float*)d_in[0];
  // dict order: x, p_valid, p_move, p_op, p_dec, p_denseop, p_densepred, start_pos
  // internal mlp order: 0 valid, 1 move, 2 op, 3 dec, 4 densepred, 5 denseop
  const int mbase[6] = {1, 7, 13, 19, 31, 25};
  const int OUTS[6] = {2, 3, 7, 2, 10, 4};

  const size_t PER_MLP = 397312;                    // bf16 elems per MLP
  const size_t PW_BYTES = 6 * PER_MLP * 2;          // 4,767,744
  const size_t CS_OFF = PW_BYTES;
  const size_t GF_OFF = CS_OFF + 2048;
  const size_t GL_OFF = GF_OFF + (size_t)NB * NS * 8 * sizeof(float);
  const size_t NEED = GL_OFF + (size_t)NB * NS * sizeof(unsigned int);
  if (ws_size < NEED) return;

  char* ws = (char*)d_ws;
  bf16* pw = (bf16*)ws;
  float* csum = (float*)(ws + CS_OFF);
  float* gatesf = (float*)(ws + GF_OFF);
  unsigned int* gfl = (unsigned int*)(ws + GL_OFF);

  for (int m = 0; m < 6; ++m) {
    const float* w1 = (const float*)d_in[mbase[m] + 0];
    const float* w2 = (const float*)d_in[mbase[m] + 2];
    const float* w3 = (const float*)d_in[mbase[m] + 4];
    bf16* dst = pw + (size_t)m * PER_MLP;
    pack_w<<<(262144 + 255) / 256, 256, 0, stream>>>(w1, dst, 512, 512, 512);
    pack_w<<<(131072 + 255) / 256, 256, 0, stream>>>(w2, dst + 262144, 512, 256, 256);
    pack_w<<<(4096 + 255) / 256, 256, 0, stream>>>(w3, dst + 393216, 256, 16, OUTS[m]);
  }
  csum_kernel<<<2, 256, 0, stream>>>((const float*)d_in[25], csum);

  GParams P;
  for (int m = 0; m < 6; ++m) {
    P.b1[m] = (const float*)d_in[mbase[m] + 1];
    P.b2[m] = (const float*)d_in[mbase[m] + 3];
    P.b3[m] = (const float*)d_in[mbase[m] + 5];
  }
  P.csum = csum; P.gatesf = gatesf; P.gflags = gfl;

  // 228 KB dynamic LDS per workgroup (CDNA5 allows up to 320 KB/WG)
  hipFuncSetAttribute((const void*)gates_kernel,
                      hipFuncAttributeMaxDynamicSharedMemorySize, SMEM_BYTES);
  gates_kernel<<<(NB * NS) / 64, 512, SMEM_BYTES, stream>>>(x, pw, P);

  hipMemsetAsync(d_out, 0, (size_t)out_size * sizeof(float), stream);
  scan_kernel<<<4, 256, 0, stream>>>(gatesf, gfl, (float*)d_out, (const int*)d_in[37]);
}
